// LSTMModel_90864328114711
// MI455X (gfx1250) — matrix operand order repaired and checked by reading, not compiled, not run on hardware
//
#include <hip/hip_runtime.h>

// ---- problem constants (match reference) ----
#define B_TOT 4096
#define T_LEN 512
#define IN_SZ 6
#define HID   32
#define NCLS  8

typedef _Float16 v16h __attribute__((ext_vector_type(16)));
typedef _Float16 v8h  __attribute__((ext_vector_type(8)));
typedef float    v8f  __attribute__((ext_vector_type(8)));

__device__ __forceinline__ v8f wmma16(v16h a, v16h b, v8f c) {
  // D = A(16x32 f16) * B(32x16 f16) + C(16x16 f32)
  return __builtin_amdgcn_wmma_f32_16x16x32_f16(
      /*neg_a=*/false, a, /*neg_b=*/false, b,
      /*c_mod=*/(short)0, c, /*reuse_a=*/false, /*reuse_b=*/false);
}

__device__ __forceinline__ float sigm(float x) {
  float e = __expf(-x);
  return __builtin_amdgcn_rcpf(1.0f + e);
}
__device__ __forceinline__ float tanh_fast(float x) {
  float e = __expf(-2.0f * x);
  return 2.0f * __builtin_amdgcn_rcpf(1.0f + e) - 1.0f;
}

// Build a B-matrix tile (32x16, f16) for gate-column tile j of weight W[128][LDK]:
// B[k][n] = W[16*j + n][k] (zero-padded when k >= LDK, e.g. LDK==6 input weights).
// Lane layout (wave32): lane = n + 16*kh, element e holds K = e + 8*kh + 8*(e>=8).
template <int LDK>
__device__ __forceinline__ v16h load_Btile(const float* __restrict__ W, int j,
                                           int n, int kh) {
  v16h b;
  const float* row = W + (size_t)(16 * j + n) * LDK;
#pragma unroll
  for (int e = 0; e < 16; e++) {
    int K = e + 8 * kh + 8 * (e >= 8);
    b[e] = (K < LDK) ? (_Float16)row[K] : (_Float16)0.0f;
  }
  return b;
}

// Read 16x32 f16 row-major tile from LDS into WMMA A-operand layout:
// lane (m + 16*kh): element e holds K = e + 8*kh + 8*(e>=8).
__device__ __forceinline__ v16h load_A(const _Float16* sh, int m, int kh) {
  v8h lo = *(const v8h*)(sh + m * 32 + 8 * kh);        // K = 8*kh .. 8*kh+7
  v8h hi = *(const v8h*)(sh + m * 32 + 16 + 8 * kh);   // K = 16+8*kh .. 23+8*kh
  v16h a;
#pragma unroll
  for (int e = 0; e < 8; e++) { a[e] = lo[e]; a[8 + e] = hi[e]; }
  return a;
}

__device__ __forceinline__ v8f splat8(float x) {
  v8f v;
#pragma unroll
  for (int r = 0; r < 8; r++) v[r] = x;
  return v;
}

__global__ __launch_bounds__(64) void lstm2_fused_kernel(
    const float* __restrict__ x,
    const float* __restrict__ W_ih0, const float* __restrict__ W_hh0,
    const float* __restrict__ b_ih0, const float* __restrict__ b_hh0,
    const float* __restrict__ W_ih1, const float* __restrict__ W_hh1,
    const float* __restrict__ b_ih1, const float* __restrict__ b_hh1,
    const float* __restrict__ W_fc,  const float* __restrict__ b_fc,
    float* __restrict__ out) {
  __shared__ _Float16 sh1[16 * 32];  // layer-0 hidden tile (row-major 16x32)
  __shared__ _Float16 sh2[16 * 32];  // layer-1 hidden tile

  const int tid  = threadIdx.x;
  const int w    = tid >> 5;   // wave id: owns hidden half [16w,16w+16)
  const int lane = tid & 31;
  const int n16  = lane & 15;  // column within 16-wide tile
  const int kh   = lane >> 4;  // K-half / M-half selector
  const int base = blockIdx.x * 16;  // 16 batch rows per workgroup

  // ---- register-resident weight B-tiles: wave w owns gate tiles j = 2*gb + w
  v16h Bih0[4], Bhh0[4], Bih1[4], Bhh1[4];
  float bias0[4], bias1[4];
#pragma unroll
  for (int gb = 0; gb < 4; gb++) {  // gb: 0=i 1=f 2=g 3=o
    int j = 2 * gb + w;
    Bih0[gb] = load_Btile<IN_SZ>(W_ih0, j, n16, kh);
    Bhh0[gb] = load_Btile<HID>(W_hh0, j, n16, kh);
    Bih1[gb] = load_Btile<HID>(W_ih1, j, n16, kh);
    Bhh1[gb] = load_Btile<HID>(W_hh1, j, n16, kh);
    int bi = 16 * j + n16;
    bias0[gb] = b_ih0[bi] + b_hh0[bi];
    bias1[gb] = b_ih1[bi] + b_hh1[bi];
  }

  v16h ah1 = {};  // h1 in A-layout (both halves, shared via LDS)
  v16h ah2 = {};  // h2 in A-layout
  v8f c1 = {};    // cell state, this wave's hidden column, rows r+8*kh
  v8f c2 = {};

  const int ncol = 16 * w + n16;  // hidden column this lane updates
  const float* xrow = x + (size_t)(base + n16) * T_LEN * IN_SZ;

  // preload x[t=0] in A layout: lanes 0-15, elements 0..5 hold K=0..5
  v16h ax = {};
  if (lane < 16) {
#pragma unroll
    for (int k = 0; k < IN_SZ; k++) ax[k] = (_Float16)xrow[k];
  }

  for (int t = 0; t < T_LEN; t++) {
    // ================= layer 0: gates = x*Wih0^T + h1*Whh0^T + b =========
    v8f gi = splat8(bias0[0]);
    v8f gf = splat8(bias0[1]);
    v8f gg = splat8(bias0[2]);
    v8f go = splat8(bias0[3]);
    gi = wmma16(ax, Bih0[0], gi); gi = wmma16(ah1, Bhh0[0], gi);
    gf = wmma16(ax, Bih0[1], gf); gf = wmma16(ah1, Bhh0[1], gf);
    gg = wmma16(ax, Bih0[2], gg); gg = wmma16(ah1, Bhh0[2], gg);
    go = wmma16(ax, Bih0[3], go); go = wmma16(ah1, Bhh0[3], go);

    // software-pipeline next x load while WMMAs drain
    v16h axn = {};
    if (t + 1 < T_LEN && lane < 16) {
      const float* xr = xrow + (size_t)(t + 1) * IN_SZ;
#pragma unroll
      for (int k = 0; k < IN_SZ; k++) axn[k] = (_Float16)xr[k];
    }

    // layer-0 state update (this wave's hidden column, 8 batch rows)
#pragma unroll
    for (int r = 0; r < 8; r++) {
      float ig = sigm(gi[r]);
      float fg = sigm(gf[r]);
      float cg = tanh_fast(gg[r]);
      float og = sigm(go[r]);
      float c  = fg * c1[r] + ig * cg;
      c1[r] = c;
      float h = og * tanh_fast(c);
      sh1[(r + 8 * kh) * 32 + ncol] = (_Float16)h;
    }
    __syncthreads();
    ah1 = load_A(sh1, n16, kh);

    // ================= layer 1: gates = h1*Wih1^T + h2*Whh1^T + b ========
    gi = splat8(bias1[0]);
    gf = splat8(bias1[1]);
    gg = splat8(bias1[2]);
    go = splat8(bias1[3]);
    gi = wmma16(ah1, Bih1[0], gi); gi = wmma16(ah2, Bhh1[0], gi);
    gf = wmma16(ah1, Bih1[1], gf); gf = wmma16(ah2, Bhh1[1], gf);
    gg = wmma16(ah1, Bih1[2], gg); gg = wmma16(ah2, Bhh1[2], gg);
    go = wmma16(ah1, Bih1[3], go); go = wmma16(ah2, Bhh1[3], go);

#pragma unroll
    for (int r = 0; r < 8; r++) {
      float ig = sigm(gi[r]);
      float fg = sigm(gf[r]);
      float cg = tanh_fast(gg[r]);
      float og = sigm(go[r]);
      float c  = fg * c2[r] + ig * cg;
      c2[r] = c;
      float h = og * tanh_fast(c);
      sh2[(r + 8 * kh) * 32 + ncol] = (_Float16)h;
    }
    __syncthreads();
    ah2 = load_A(sh2, n16, kh);

    ax = axn;
  }

  // ================= final FC: out = h2_last @ W_fc^T + b_fc =============
  // sh2 holds the final 16x32 h2 tile (synced above). 128 outputs / 64 thr.
  for (int idx = tid; idx < 16 * NCLS; idx += 64) {
    int m = idx & 15;
    int cls = idx >> 4;
    float s = b_fc[cls];
#pragma unroll
    for (int n = 0; n < HID; n++)
      s += (float)sh2[m * 32 + n] * W_fc[cls * HID + n];
    out[(size_t)(base + m) * NCLS + cls] = s;
  }
}

extern "C" void kernel_launch(void* const* d_in, const int* in_sizes, int n_in,
                              void* d_out, int out_size, void* d_ws,
                              size_t ws_size, hipStream_t stream) {
  (void)in_sizes; (void)n_in; (void)out_size; (void)d_ws; (void)ws_size;
  const float* x     = (const float*)d_in[0];
  const float* W_ih0 = (const float*)d_in[1];
  const float* W_hh0 = (const float*)d_in[2];
  const float* b_ih0 = (const float*)d_in[3];
  const float* b_hh0 = (const float*)d_in[4];
  const float* W_ih1 = (const float*)d_in[5];
  const float* W_hh1 = (const float*)d_in[6];
  const float* b_ih1 = (const float*)d_in[7];
  const float* b_hh1 = (const float*)d_in[8];
  const float* W_fc  = (const float*)d_in[9];
  const float* b_fc  = (const float*)d_in[10];
  float* out = (float*)d_out;

  dim3 grid(B_TOT / 16);  // 256 workgroups
  dim3 block(64);         // 2 waves: hidden halves
  lstm2_fused_kernel<<<grid, block, 0, stream>>>(
      x, W_ih0, W_hh0, b_ih0, b_hh0, W_ih1, W_hh1, b_ih1, b_hh1, W_fc, b_fc,
      out);
}
